// get_model_15736760173078
// MI455X (gfx1250) — matrix operand order repaired and checked by reading, not compiled, not run on hardware
//
#include <hip/hip_runtime.h>
#include <hip/hip_bf16.h>

typedef __attribute__((ext_vector_type(16))) _Float16 v16h;
typedef __attribute__((ext_vector_type(8)))  _Float16 v8h;
typedef __attribute__((ext_vector_type(8)))  float    v8f;

#define BN_EPS 1e-5f

// ---------------------------------------------------------------- transpose
// xyz (B,3,N) f32 -> pts (B,N,3) f32
__global__ void k_transpose(const float* __restrict__ xyz, float* __restrict__ pts,
                            int Bn, int N) {
  int t = blockIdx.x * blockDim.x + threadIdx.x;
  int total = Bn * N * 3;
  if (t >= total) return;
  int c = t % 3;
  int n = (t / 3) % N;
  int b = t / (3 * N);
  pts[t] = xyz[((size_t)b * 3 + c) * N + n];
}

// ---------------------------------------------------------------- FPS
// one block (256 threads) per batch; Np <= 4096
__global__ void k_fps(const float* __restrict__ pts, int* __restrict__ fidx,
                      float* __restrict__ new_xyz, int Np, int npoint) {
  const int b   = blockIdx.x;
  const int tid = threadIdx.x;
  const int NT  = 256;
  __shared__ float sdist[4096];
  __shared__ float rv[256];
  __shared__ int   ri[256];
  __shared__ int   sfar;
  __shared__ float scx, scy, scz;

  for (int i = tid; i < Np; i += NT) sdist[i] = 1e10f;
  if (tid == 0) sfar = 0;
  __syncthreads();

  const float* P = pts + (size_t)b * Np * 3;
  for (int s = 0; s < npoint; ++s) {
    if (tid == 0) {
      int f = sfar;
      fidx[b * npoint + s] = f;
      float cx = P[f * 3 + 0], cy = P[f * 3 + 1], cz = P[f * 3 + 2];
      scx = cx; scy = cy; scz = cz;
      new_xyz[((size_t)b * npoint + s) * 3 + 0] = cx;
      new_xyz[((size_t)b * npoint + s) * 3 + 1] = cy;
      new_xyz[((size_t)b * npoint + s) * 3 + 2] = cz;
    }
    __syncthreads();
    float cx = scx, cy = scy, cz = scz;
    float bv = -1.0f; int bi = 0;
    for (int i = tid; i < Np; i += NT) {
      float dx = P[i * 3 + 0] - cx;
      float dy = P[i * 3 + 1] - cy;
      float dz = P[i * 3 + 2] - cz;
      float d  = dx * dx + dy * dy + dz * dz;
      float nd = fminf(sdist[i], d);
      sdist[i] = nd;
      if (nd > bv) { bv = nd; bi = i; }   // ascending i -> lowest-index tie kept
    }
    rv[tid] = bv; ri[tid] = bi;
    __syncthreads();
    for (int off = NT / 2; off > 0; off >>= 1) {
      if (tid < off) {
        float v2 = rv[tid + off]; int i2 = ri[tid + off];
        if (v2 > rv[tid] || (v2 == rv[tid] && i2 < ri[tid])) { rv[tid] = v2; ri[tid] = i2; }
      }
      __syncthreads();
    }
    if (tid == 0) sfar = ri[0];
    __syncthreads();
  }
}

// ---------------------------------------------------------------- ball query
// thread per (b,s): first ns indices (ascending) with d2 <= r2, pad with first
__global__ void k_ballquery(const float* __restrict__ pts, const float* __restrict__ q,
                            int* __restrict__ idx, int Bn, int Np, int S, int ns, float r2) {
  int t = blockIdx.x * blockDim.x + threadIdx.x;
  if (t >= Bn * S) return;
  int b = t / S;
  const float* P = pts + (size_t)b * Np * 3;
  float qx = q[(size_t)t * 3 + 0], qy = q[(size_t)t * 3 + 1], qz = q[(size_t)t * 3 + 2];
  int* out = idx + (size_t)t * ns;
  int cnt = 0, first = 0;
  for (int n = 0; n < Np && cnt < ns; ++n) {
    float dx = P[n * 3 + 0] - qx;
    float dy = P[n * 3 + 1] - qy;
    float dz = P[n * 3 + 2] - qz;
    float d  = dx * dx + dy * dy + dz * dz;
    if (d <= r2) { if (cnt == 0) first = n; out[cnt++] = n; }
  }
  for (; cnt < ns; ++cnt) out[cnt] = first;
}

// ---------------------------------------------------------------- grouping
// thread per (b,s,j): row = [xyz(gi)-q(s), feats(gi), zeros...] of length ld
__global__ void k_group(const float* __restrict__ pts, const float* __restrict__ q,
                        const float* __restrict__ feats, const int* __restrict__ idx,
                        _Float16* __restrict__ X, int Bn, int Np, int S, int ns,
                        int Cf, int ld) {
  int t = blockIdx.x * blockDim.x + threadIdx.x;
  int total = Bn * S * ns;
  if (t >= total) return;
  int b  = t / (S * ns);
  int s  = (t / ns) % S;
  int gi = idx[t];
  const float* p  = pts + ((size_t)b * Np + gi) * 3;
  const float* qq = q   + ((size_t)b * S  + s ) * 3;
  _Float16* row = X + (size_t)t * ld;
  row[0] = (_Float16)(p[0] - qq[0]);
  row[1] = (_Float16)(p[1] - qq[1]);
  row[2] = (_Float16)(p[2] - qq[2]);
  if (Cf > 0) {
    const float* f = feats + ((size_t)b * Np + gi) * Cf;
    for (int c = 0; c < Cf; ++c) row[3 + c] = (_Float16)f[c];
  }
  for (int c = 3 + Cf; c < ld; ++c) row[c] = (_Float16)0.f;   // zero K-pad
}

// set_abstraction_all: row per (b,k) = [xyz raw, feats, zeros] of length ld
__global__ void k_concat_all(const float* __restrict__ xyz, const float* __restrict__ feats,
                             _Float16* __restrict__ X, int Bn, int K, int Cf, int ld) {
  int t = blockIdx.x * blockDim.x + threadIdx.x;
  if (t >= Bn * K) return;
  _Float16* row = X + (size_t)t * ld;
  row[0] = (_Float16)xyz[(size_t)t * 3 + 0];
  row[1] = (_Float16)xyz[(size_t)t * 3 + 1];
  row[2] = (_Float16)xyz[(size_t)t * 3 + 2];
  const float* f = feats + (size_t)t * Cf;
  for (int c = 0; c < Cf; ++c) row[3 + c] = (_Float16)f[c];
  for (int c = 3 + Cf; c < ld; ++c) row[c] = (_Float16)0.f;
}

// head concat: [l3_pts(1024), text_emb(384)] per batch row (1408 = 44*32, no pad)
__global__ void k_concat_head(const float* __restrict__ l3, const float* __restrict__ te,
                              _Float16* __restrict__ X, int Bn) {
  int t = blockIdx.x * blockDim.x + threadIdx.x;
  if (t >= Bn * 1408) return;
  int b = t / 1408, c = t % 1408;
  X[t] = (_Float16)(c < 1024 ? l3[(size_t)b * 1024 + c] : te[(size_t)b * 384 + (c - 1024)]);
}

// ---------------------------------------------------------------- weight pack
// W (O,C) f32 -> Wp (Opad, ldk) f16, zero-filled pad rows/cols
__global__ void k_packw(const float* __restrict__ W, _Float16* __restrict__ Wp,
                        int O, int C, int Opad, int ldk) {
  int t = blockIdx.x * blockDim.x + threadIdx.x;
  if (t >= Opad * ldk) return;
  int n = t / ldk, k = t % ldk;
  Wp[t] = (n < O && k < C) ? (_Float16)W[(size_t)n * C + k] : (_Float16)0.f;
}

// ---------------------------------------------------------------- WMMA GEMM (small)
// Y[M,O] = X[M,ldk](f16) * Wp[Opad,ldk]^T(f16) + bias[O]; one 16x16 tile per wave.
__global__ void k_wmma_gemm(const _Float16* __restrict__ X, const _Float16* __restrict__ Wp,
                            const float* __restrict__ bias, float* __restrict__ Y,
                            int M, int ldk, int O) {
  const int lane  = threadIdx.x & 31;
  const int wave  = threadIdx.x >> 5;
  const int mbase = (blockIdx.y * 4 + wave) * 16;
  const int nbase = blockIdx.x * 16;
  if (mbase >= M) return;                 // wave-uniform exit; EXEC stays all-1s
  const int l16  = lane & 15;
  const int hsel = lane >> 4;
  // A layout (16-bit A, 16x32): lane<16 -> K {0..7}U{16..23}; lane>=16 -> +8
  const _Float16* xr = X  + (size_t)(mbase + l16) * ldk + hsel * 8;
  // B layout (32x16): n = lane%16, K same as A: lane<16 -> {0..7}U{16..23}; lane>=16 -> +8
  const _Float16* wr = Wp + (size_t)(nbase + l16) * ldk + hsel * 8;

  v8f acc = {};
#pragma unroll 2
  for (int k0 = 0; k0 < ldk; k0 += 32) {
    v8h a0 = *(const v8h*)(xr);
    v8h a1 = *(const v8h*)(xr + 16);
    v8h b0 = *(const v8h*)(wr);
    v8h b1 = *(const v8h*)(wr + 16);
    v16h a = __builtin_shufflevector(a0, a1, 0,1,2,3,4,5,6,7,8,9,10,11,12,13,14,15);
    v16h b = __builtin_shufflevector(b0, b1, 0,1,2,3,4,5,6,7,8,9,10,11,12,13,14,15);
    acc = __builtin_amdgcn_wmma_f32_16x16x32_f16(false, a, false, b,
                                                 (short)0, acc, false, false);
    xr += 32;
    wr += 32;
  }
#pragma unroll
  for (int i = 0; i < 8; ++i) {
    int mm = mbase + i + 8 * hsel;
    int nn = nbase + l16;
    if (nn < O) Y[(size_t)mm * O + nn] = acc[i] + bias[nn];
  }
}

// ------------------------------------------------------- WMMA GEMM (blocked)
// Each wave owns a 64x32 output block: 4 M-tiles x 2 N-tiles, 8 WMMAs per K-step.
// Requires M % 256 == 0 and O % 32 == 0 (checked by caller).
__global__ void k_wmma_gemm_blk(const _Float16* __restrict__ X, const _Float16* __restrict__ Wp,
                                const float* __restrict__ bias, float* __restrict__ Y,
                                int M, int ldk, int O) {
  const int lane  = threadIdx.x & 31;
  const int wave  = threadIdx.x >> 5;
  const int mbase = (blockIdx.y * 4 + wave) * 64;
  const int nbase = blockIdx.x * 32;
  const int l16   = lane & 15;
  const int hsel  = lane >> 4;

  const _Float16* xr = X  + (size_t)(mbase + l16) * ldk + hsel * 8;
  const _Float16* wr = Wp + (size_t)(nbase + l16) * ldk + hsel * 8;
  const size_t mstep = (size_t)16 * ldk;

  v8f acc[4][2] = {};
  for (int k0 = 0; k0 < ldk; k0 += 32) {
    v16h a[4], b[2];
#pragma unroll
    for (int t = 0; t < 4; ++t) {
      const _Float16* p = xr + (size_t)t * mstep;
      v8h lo = *(const v8h*)(p);
      v8h hi = *(const v8h*)(p + 16);
      a[t] = __builtin_shufflevector(lo, hi, 0,1,2,3,4,5,6,7,8,9,10,11,12,13,14,15);
    }
#pragma unroll
    for (int u = 0; u < 2; ++u) {
      const _Float16* p = wr + (size_t)u * mstep;
      v8h lo = *(const v8h*)(p);
      v8h hi = *(const v8h*)(p + 16);
      b[u] = __builtin_shufflevector(lo, hi, 0,1,2,3,4,5,6,7,8,9,10,11,12,13,14,15);
    }
#pragma unroll
    for (int t = 0; t < 4; ++t)
#pragma unroll
      for (int u = 0; u < 2; ++u)
        acc[t][u] = __builtin_amdgcn_wmma_f32_16x16x32_f16(false, a[t], false, b[u],
                                                           (short)0, acc[t][u], false, false);
    xr += 32;
    wr += 32;
  }
#pragma unroll
  for (int t = 0; t < 4; ++t) {
#pragma unroll
    for (int u = 0; u < 2; ++u) {
      int nn = nbase + 16 * u + l16;
      float bv = bias[nn];
#pragma unroll
      for (int i = 0; i < 8; ++i) {
        int mm = mbase + 16 * t + i + 8 * hsel;
        Y[(size_t)mm * O + nn] = acc[t][u][i] + bv;
      }
    }
  }
}

// ---------------------------------------------------------------- batch-norm
__global__ void k_zero(float* __restrict__ p, int n) {
  int t = blockIdx.x * blockDim.x + threadIdx.x;
  if (t < n) p[t] = 0.f;
}

// per-block row chunk, threads stride channels (coalesced), atomics to global sums
__global__ void k_bn_partial(const float* __restrict__ Y, float* __restrict__ sums,
                             int M, int O, int rowsPerBlock) {
  int r0 = blockIdx.x * rowsPerBlock;
  int r1 = r0 + rowsPerBlock; if (r1 > M) r1 = M;
  for (int c = threadIdx.x; c < O; c += blockDim.x) {
    float s = 0.f, q = 0.f;
    for (int r = r0; r < r1; ++r) {
      float v = Y[(size_t)r * O + c];
      s += v; q += v * v;
    }
    atomicAdd(&sums[c], s);
    atomicAdd(&sums[O + c], q);
  }
}

__global__ void k_bn_finalize(const float* __restrict__ sums, const float* __restrict__ g,
                              const float* __restrict__ beta, float* __restrict__ scale,
                              float* __restrict__ shift, int M, int O) {
  int c = blockIdx.x * blockDim.x + threadIdx.x;
  if (c >= O) return;
  float mean = sums[c] / (float)M;
  float var  = sums[O + c] / (float)M - mean * mean;
  float sc   = g[c] * rsqrtf(var + BN_EPS);
  scale[c] = sc;
  shift[c] = beta[c] - mean * sc;
}

__global__ void k_bn_relu(const float* __restrict__ Y, const float* __restrict__ scale,
                          const float* __restrict__ shift, _Float16* __restrict__ X,
                          int M, int O) {
  int t = blockIdx.x * blockDim.x + threadIdx.x;
  if (t >= M * O) return;
  int c = t % O;
  float v = Y[t] * scale[c] + shift[c];
  X[t] = (_Float16)fmaxf(v, 0.f);
}

// ---------------------------------------------------------------- max pool
// X (BS*K, O) f16 -> out (BS, O) f32, max over K
__global__ void k_maxpool(const _Float16* __restrict__ X, float* __restrict__ out,
                          int BS, int K, int O) {
  int t = blockIdx.x * blockDim.x + threadIdx.x;
  if (t >= BS * O) return;
  int bs = t / O, c = t % O;
  const _Float16* p = X + (size_t)bs * K * O + c;
  float m = -1e30f;
  for (int j = 0; j < K; ++j) m = fmaxf(m, (float)p[(size_t)j * O]);
  out[t] = m;
}

// ---------------------------------------------------------------- host
static inline size_t align_up(size_t x) { return (x + 255) & ~(size_t)255; }

extern "C" void kernel_launch(void* const* d_in, const int* in_sizes, int n_in,
                              void* d_out, int out_size, void* d_ws, size_t ws_size,
                              hipStream_t stream) {
  (void)in_sizes; (void)n_in; (void)out_size; (void)ws_size;
  const float* xyz  = (const float*)d_in[0];
  const float* temb = (const float*)d_in[1];
  auto F = [&](int i) { return (const float*)d_in[i]; };

  char* ws = (char*)d_ws;
  size_t off = 0;
  auto alloc = [&](size_t bytes) { void* p = ws + off; off = align_up(off + bytes); return p; };

  const int Bn = 32, N = 4096;
  float* pts    = (float*)alloc((size_t)Bn * N * 3 * 4);
  int*   fidx   = (int*)  alloc((size_t)Bn * 512 * 4);
  float* nx1    = (float*)alloc((size_t)Bn * 512 * 3 * 4);
  float* nx2    = (float*)alloc((size_t)Bn * 128 * 3 * 4);
  int*   idx1   = (int*)  alloc((size_t)Bn * 512 * 32 * 4);
  int*   idx2   = (int*)  alloc((size_t)Bn * 128 * 64 * 4);
  float* l1p    = (float*)alloc((size_t)Bn * 512 * 128 * 4);
  float* l2p    = (float*)alloc((size_t)Bn * 128 * 256 * 4);
  float* bnsums = (float*)alloc(2 * 1024 * 4);
  float* bnscal = (float*)alloc(1024 * 4);
  float* bnshft = (float*)alloc(1024 * 4);
  _Float16* Wp  = (_Float16*)alloc((size_t)1048576 * 2);   // max Opad*ldk (512*1408)
  _Float16* Xb  = (_Float16*)alloc((size_t)41943040 * 2);  // max rows*ldk (262144*160)
  float*    Yb  = (float*)   alloc((size_t)67108864 * 4);  // max rows*O   (524288*128)

  float* outx = (float*)d_out;       // (32,7)
  float* l3p  = outx + 32 * 7;       // (32,1024)

  auto gemm = [&](const _Float16* X, int pi, float* Y, int M, int C, int ldk, int O) {
    int Opad = (O + 15) & ~15;
    k_packw<<<((size_t)Opad * ldk + 255) / 256, 256, 0, stream>>>(F(pi), Wp, O, C, Opad, ldk);
    if ((M % 256) == 0 && (O % 32) == 0) {
      dim3 grid(O / 32, M / 256);
      k_wmma_gemm_blk<<<grid, 128, 0, stream>>>(X, Wp, F(pi + 1), Y, M, ldk, O);
    } else {
      dim3 grid(Opad / 16, (M + 63) / 64);
      k_wmma_gemm<<<grid, 128, 0, stream>>>(X, Wp, F(pi + 1), Y, M, ldk, O);
    }
  };
  auto bnrelu = [&](const float* Y, int pi, _Float16* Xn, int M, int O) {
    k_zero<<<(2 * O + 255) / 256, 256, 0, stream>>>(bnsums, 2 * O);
    int rpb = 256;
    k_bn_partial<<<(M + rpb - 1) / rpb, 256, 0, stream>>>(Y, bnsums, M, O, rpb);
    k_bn_finalize<<<(O + 255) / 256, 256, 0, stream>>>(bnsums, F(pi + 2), F(pi + 3),
                                                       bnscal, bnshft, M, O);
    size_t tot = (size_t)M * O;
    k_bn_relu<<<(tot + 255) / 256, 256, 0, stream>>>(Y, bnscal, bnshft, Xn, M, O);
  };
  // every BN output O is a multiple of 32, so next layer's ldk == O (no re-pad)
  auto layer = [&](int pi, int M, int C, int ldk, int O) {
    gemm(Xb, pi, Yb, M, C, ldk, O);
    bnrelu(Yb, pi, Xb, M, O);
  };

  // ---- stage 0: transpose (B,3,N) -> (B,N,3)
  k_transpose<<<(Bn * N * 3 + 255) / 256, 256, 0, stream>>>(xyz, pts, Bn, N);

  // ---- SA1: 512 centers, r=0.2, ns=32, MLP 3->64->64->128
  k_fps<<<Bn, 256, 0, stream>>>(pts, fidx, nx1, N, 512);
  k_ballquery<<<(Bn * 512 + 255) / 256, 256, 0, stream>>>(pts, nx1, idx1, Bn, N, 512, 32, 0.04f);
  k_group<<<(Bn * 512 * 32 + 255) / 256, 256, 0, stream>>>(pts, nx1, nullptr, idx1, Xb,
                                                           Bn, N, 512, 32, 0, 32);
  const int M1 = Bn * 512 * 32;
  layer(2,  M1, 3,  32, 64);
  layer(6,  M1, 64, 64, 64);
  layer(10, M1, 64, 64, 128);
  k_maxpool<<<(Bn * 512 * 128 + 255) / 256, 256, 0, stream>>>(Xb, l1p, Bn * 512, 32, 128);

  // ---- SA2: 128 centers, r=0.4, ns=64, MLP 131->128->128->256
  k_fps<<<Bn, 256, 0, stream>>>(nx1, fidx, nx2, 512, 128);
  k_ballquery<<<(Bn * 128 + 255) / 256, 256, 0, stream>>>(nx1, nx2, idx2, Bn, 512, 128, 64, 0.16f);
  k_group<<<(Bn * 128 * 64 + 255) / 256, 256, 0, stream>>>(nx1, nx2, l1p, idx2, Xb,
                                                           Bn, 512, 128, 64, 128, 160);
  const int M2 = Bn * 128 * 64;
  layer(14, M2, 131, 160, 128);
  layer(18, M2, 128, 128, 128);
  layer(22, M2, 128, 128, 256);
  k_maxpool<<<(Bn * 128 * 256 + 255) / 256, 256, 0, stream>>>(Xb, l2p, Bn * 128, 64, 256);

  // ---- SA3 (group-all): MLP 259->256->512->1024, pool over 128
  k_concat_all<<<(Bn * 128 + 255) / 256, 256, 0, stream>>>(nx2, l2p, Xb, Bn, 128, 256, 288);
  const int M3 = Bn * 128;
  layer(26, M3, 259, 288, 256);
  layer(30, M3, 256, 256, 512);
  layer(34, M3, 512, 512, 1024);
  k_maxpool<<<(Bn * 1024 + 255) / 256, 256, 0, stream>>>(Xb, l3p, Bn, 128, 1024);

  // ---- head: [l3,text] 1408 -> 512 -> 256 -> 7
  k_concat_head<<<(Bn * 1408 + 255) / 256, 256, 0, stream>>>(l3p, temb, Xb, Bn);
  layer(38, Bn, 1408, 1408, 512);
  layer(42, Bn, 512,  512,  256);
  gemm(Xb, 46, outx, Bn, 256, 256, 7);   // final linear, no BN/ReLU, straight to d_out
}
